// SparseAutoencoder_27556510171676
// MI455X (gfx1250) — matrix operand order repaired and checked by reading, not compiled, not run on hardware
//
#include <hip/hip_runtime.h>

// ---------------------------------------------------------------------------
// Sparse autoencoder forward for MI455X (gfx1250, wave32, WMMA).
// B=8192, D_IN=1024, D_HID=2048, N_FEAT=16384, K(topk)=64
// ---------------------------------------------------------------------------

// Toggle for the async global->LDS staging path in the top-k kernel.
#define USE_ASYNC_LDS 1

typedef __attribute__((ext_vector_type(16))) __bf16 v16bf;
typedef __attribute__((ext_vector_type(8)))  __bf16 v8bf;
typedef __attribute__((ext_vector_type(8)))  float  v8f;

union BF16Frag { v16bf v; v8bf h[2]; };

__device__ __forceinline__ unsigned short f32_to_bf16(float f) {
  unsigned int u = __float_as_uint(f);
  unsigned int r = u + 0x7FFFu + ((u >> 16) & 1u);   // round-to-nearest-even
  return (unsigned short)(r >> 16);
}

// ---------------------------------------------------------------------------
// fp32 -> bf16 conversion (vectorized x4)
// ---------------------------------------------------------------------------
__global__ void cvt_f32_to_bf16(const float* __restrict__ in,
                                unsigned short* __restrict__ out, int n) {
  int i = (blockIdx.x * blockDim.x + threadIdx.x) * 4;
  if (i + 3 < n) {
    float4 v = *(const float4*)(in + i);
    out[i + 0] = f32_to_bf16(v.x);
    out[i + 1] = f32_to_bf16(v.y);
    out[i + 2] = f32_to_bf16(v.z);
    out[i + 3] = f32_to_bf16(v.w);
  } else {
    for (; i < n; ++i) out[i] = f32_to_bf16(in[i]);
  }
}

// ---------------------------------------------------------------------------
// bf16 WMMA GEMM:  out[M,N] = A[M,K] @ Bm[N,K]^T  (+ epilogue)
//   EPI==1:  out_bf16 = bf16( relu(acc + bias1[n]) )        (encoder layer 1)
//   EPI==2:  out_f32  = (acc + bias1[n])*scale[n] + bias2[n] (encoder layer 2)
//            stored non-temporally (537MB stream must not evict L2-resident
//            weights/activations: hidb 33.5MB + W2b 67MB << 192MB L2).
// Block: 256 threads = 8 waves; wave grid 4(M) x 2(N); each wave 32x32 tile
// via 2x2 v_wmma_f32_16x16x32_bf16. Block tile = 128x64.
// ---------------------------------------------------------------------------
template <int EPI>
__global__ __launch_bounds__(256)
void gemm_bf16_wmma(const unsigned short* __restrict__ A,
                    const unsigned short* __restrict__ Bm,
                    const float* __restrict__ bias1,
                    const float* __restrict__ scale,
                    const float* __restrict__ bias2,
                    unsigned short* __restrict__ outBf,
                    float* __restrict__ outF,
                    int M, int N, int K) {
  const int lane  = threadIdx.x & 31;
  const int wid   = threadIdx.x >> 5;
  const int waveM = wid & 3;
  const int waveN = wid >> 2;
  const int bM    = blockIdx.x * 128;
  const int bN    = blockIdx.y * 64;
  const int l15   = lane & 15;
  const int lhi   = (lane >> 4) & 1;

  // ISA fragment addressing (CDNA5 ISA §7.12.2):
  // A 16x32 bf16: lane holds row (lane&15); K-halves at kbA = 8*(lane>>4),
  //               elements 0..7 = K[k0+kbA .. +7], 8..15 = K[k0+16+kbA .. +7]
  // B 32x16 bf16 (from row-major [N,K] weights): lane holds col (lane&15),
  //               same K order as A: kbB = 8*(lane>>4), K[k0+kbB..+7], K[k0+16+kbB..+7]
  const int kbA = lhi * 8;
  const int kbB = lhi * 8;
  const size_t rowA0 = (size_t)(bM + waveM * 32 + l15);
  const size_t colB0 = (size_t)(bN + waveN * 32 + l15);

  v8f acc[2][2] = {};

  for (int k0 = 0; k0 < K; k0 += 32) {
    BF16Frag a[2], b[2];
#pragma unroll
    for (int mt = 0; mt < 2; ++mt) {
      const unsigned short* p = A + (rowA0 + mt * 16) * K + k0 + kbA;
      a[mt].h[0] = *(const v8bf*)(p);
      a[mt].h[1] = *(const v8bf*)(p + 16);
      // Unconditional speculative prefetch of next K-tile (ISA §10.5:
      // invalid speculative prefetch addresses are silently dropped).
      __builtin_prefetch(p + 32, 0, 3);
    }
#pragma unroll
    for (int nt = 0; nt < 2; ++nt) {
      const unsigned short* p = Bm + (colB0 + nt * 16) * K + k0 + kbB;
      b[nt].h[0] = *(const v8bf*)(p);
      b[nt].h[1] = *(const v8bf*)(p + 16);
      __builtin_prefetch(p + 32, 0, 3);
    }
#pragma unroll
    for (int mt = 0; mt < 2; ++mt)
#pragma unroll
      for (int nt = 0; nt < 2; ++nt)
        acc[mt][nt] = __builtin_amdgcn_wmma_f32_16x16x32_bf16(
            false, a[mt].v, false, b[nt].v, (short)0, acc[mt][nt], false, false);
  }

  // Epilogue. C/D layout: VGPR v -> row = v + 8*(lane>>4), col = lane&15.
#pragma unroll
  for (int mt = 0; mt < 2; ++mt) {
#pragma unroll
    for (int nt = 0; nt < 2; ++nt) {
      const int gn = bN + waveN * 32 + nt * 16 + l15;
#pragma unroll
      for (int v = 0; v < 8; ++v) {
        const int gm = bM + waveM * 32 + mt * 16 + v + 8 * lhi;
        float val = acc[mt][nt][v];
        if (EPI == 1) {
          val += bias1[gn];
          val = val > 0.f ? val : 0.f;
          outBf[(size_t)gm * N + gn] = f32_to_bf16(val);
        } else {
          val = (val + bias1[gn]) * scale[gn] + bias2[gn];
          __builtin_nontemporal_store(val, &outF[(size_t)gm * N + gn]);
        }
      }
    }
  }
}

// ---------------------------------------------------------------------------
// 32x32 tiled transpose: in[rows,cols] -> out[cols,rows]  (Wd -> Wd^T)
// ---------------------------------------------------------------------------
__global__ __launch_bounds__(256)
void transpose_kernel(const float* __restrict__ in, float* __restrict__ out,
                      int rows, int cols) {
  __shared__ float tile[32][33];
  const int c0 = blockIdx.x * 32, r0 = blockIdx.y * 32;
  const int tx = threadIdx.x & 31, ty = threadIdx.x >> 5;  // 8 row-passes
  for (int i = ty; i < 32; i += 8)
    tile[i][tx] = in[(size_t)(r0 + i) * cols + c0 + tx];
  __syncthreads();
  for (int i = ty; i < 32; i += 8)
    out[(size_t)(c0 + i) * rows + r0 + tx] = tile[tx][i];
}

// ---------------------------------------------------------------------------
// Per-row top-k (k=64 of nfeat=16384), row resident in LDS (64KB of the
// 320KB/WGP). Row is staged global->LDS with GLOBAL_LOAD_ASYNC_TO_LDS_B128
// (ASYNCcnt-tracked, no VGPR bounce), then 64 x (strided argmax + tree
// reduce + mask). Then zero-fill the feats row (non-temporal: 537MB stream)
// and scatter relu(topk). Also emits compact (idx,val) for the decoder.
// Dynamic LDS: nfeat*4 + blockDim*8 + k*8 bytes. NOTE: this kernel has no
// static LDS, so `vals[i]` lives at LDS byte offset i*4 exactly.
// ---------------------------------------------------------------------------
__global__ void topk_scatter_kernel(const float* __restrict__ h,
                                    float* __restrict__ feats,
                                    int* __restrict__ cidx,
                                    float* __restrict__ cval,
                                    int nfeat, int k) {
  extern __shared__ unsigned char smraw[];
  float* vals = (float*)smraw;                    // [nfeat]  (LDS offset 0)
  float* rmax = vals + nfeat;                     // [blockDim.x]
  int*   ridx = (int*)(rmax + blockDim.x);        // [blockDim.x]
  float* tv   = (float*)(ridx + blockDim.x);      // [k]
  int*   ti   = (int*)(tv + k);                   // [k]

  const int row = blockIdx.x;
  const float* hrow = h + (size_t)row * nfeat;

#if USE_ASYNC_LDS
  // Stage the 64KB row into LDS via async copy: nfeat/4 x 16B chunks.
  for (int c = threadIdx.x; c < (nfeat >> 2); c += blockDim.x) {
    unsigned ldsOff = (unsigned)(c * 16);          // byte offset of vals[c*4]
    const float* g = hrow + c * 4;
    asm volatile("global_load_async_to_lds_b128 %0, %1, off"
                 :: "v"(ldsOff), "v"(g)
                 : "memory");
  }
  asm volatile("s_wait_asynccnt 0" ::: "memory");  // this wave's copies done
#else
  for (int i = threadIdx.x; i < nfeat; i += blockDim.x) vals[i] = hrow[i];
#endif
  __syncthreads();                                 // all waves' copies visible

  for (int it = 0; it < k; ++it) {
    float m = -3.402823466e38f; int mi = 0;
    for (int i = threadIdx.x; i < nfeat; i += blockDim.x) {
      float v = vals[i];
      if (v > m) { m = v; mi = i; }
    }
    rmax[threadIdx.x] = m; ridx[threadIdx.x] = mi;
    __syncthreads();
    for (int s = blockDim.x >> 1; s > 0; s >>= 1) {
      if ((int)threadIdx.x < s) {
        if (rmax[threadIdx.x + s] > rmax[threadIdx.x]) {
          rmax[threadIdx.x] = rmax[threadIdx.x + s];
          ridx[threadIdx.x] = ridx[threadIdx.x + s];
        }
      }
      __syncthreads();
    }
    if (threadIdx.x == 0) {
      tv[it] = rmax[0]; ti[it] = ridx[0];
      vals[ridx[0]] = -3.402823466e38f;   // mask winner
    }
    __syncthreads();
  }

  float* frow = feats + (size_t)row * nfeat;
  for (int i = threadIdx.x; i < nfeat; i += blockDim.x)
    __builtin_nontemporal_store(0.f, &frow[i]);
  __syncthreads();
  if ((int)threadIdx.x < k) {
    float v = tv[threadIdx.x];
    v = v > 0.f ? v : 0.f;                // relu
    int id = ti[threadIdx.x];
    frow[id] = v;
    cidx[row * k + threadIdx.x] = id;
    cval[row * k + threadIdx.x] = v;
  }
}

// ---------------------------------------------------------------------------
// Sparse decoder: recon[row,:] = bd + sum_j cval[row,j] * WdT[cidx[row,j],:]
// One block (256 thr) per row; 4 outputs/thread via float4. 1.1 GF vs 275 GF
// dense — pure bandwidth (~2 GB gathered; WdT 67MB stays L2-resident).
// ---------------------------------------------------------------------------
__global__ __launch_bounds__(256)
void sparse_decoder_kernel(const float* __restrict__ WdT,
                           const float* __restrict__ bd,
                           const int* __restrict__ cidx,
                           const float* __restrict__ cval,
                           float* __restrict__ recon, int din, int k) {
  __shared__ int   si[64];
  __shared__ float sv[64];
  const int row = blockIdx.x;
  if ((int)threadIdx.x < k) {
    si[threadIdx.x] = cidx[row * k + threadIdx.x];
    sv[threadIdx.x] = cval[row * k + threadIdx.x];
  }
  __syncthreads();
  const int d = threadIdx.x * 4;
  float4 acc = *(const float4*)(bd + d);
  for (int j = 0; j < k; ++j) {
    const float v = sv[j];
    const float4 w = *(const float4*)(WdT + (size_t)si[j] * din + d);
    acc.x += v * w.x; acc.y += v * w.y; acc.z += v * w.z; acc.w += v * w.w;
  }
  *(float4*)(recon + (size_t)row * din + d) = acc;
}

// ---------------------------------------------------------------------------
// Host-side orchestration
// ---------------------------------------------------------------------------
extern "C" void kernel_launch(void* const* d_in, const int* in_sizes, int n_in,
                              void* d_out, int out_size, void* d_ws, size_t ws_size,
                              hipStream_t stream) {
  constexpr int B_   = 8192;
  constexpr int DIN  = 1024;
  constexpr int DHID = 2048;
  constexpr int NF   = 16384;
  constexpr int TK   = 64;

  const float* x     = (const float*)d_in[0];
  const float* W1    = (const float*)d_in[1];
  const float* b1    = (const float*)d_in[2];
  const float* W2    = (const float*)d_in[3];
  const float* b2    = (const float*)d_in[4];
  const float* scale = (const float*)d_in[5];
  const float* bias  = (const float*)d_in[6];
  const float* Wd    = (const float*)d_in[7];
  const float* bd    = (const float*)d_in[8];
  (void)in_sizes; (void)n_in; (void)out_size; (void)ws_size;

  // d_out: recon [B, DIN] then feats [B, NF], fp32
  float* recon = (float*)d_out;
  float* feats = (float*)d_out + (size_t)B_ * DIN;

  // Workspace carve-up (256B aligned)
  char* ws = (char*)d_ws;
  size_t off = 0;
  auto carve = [&](size_t bytes) -> void* {
    void* p = ws + off;
    off = (off + bytes + 255) & ~(size_t)255;
    return p;
  };
  unsigned short* xb   = (unsigned short*)carve((size_t)B_ * DIN * 2);
  unsigned short* W1b  = (unsigned short*)carve((size_t)DHID * DIN * 2);
  unsigned short* W2b  = (unsigned short*)carve((size_t)NF * DHID * 2);
  unsigned short* hidb = (unsigned short*)carve((size_t)B_ * DHID * 2);
  float*          hbuf = (float*)carve((size_t)B_ * NF * 4);
  float*          WdT  = (float*)carve((size_t)NF * DIN * 4);
  int*            cidx = (int*)carve((size_t)B_ * TK * 4);
  float*          cval = (float*)carve((size_t)B_ * TK * 4);

  // 1) fp32 -> bf16 conversions
  {
    int n;
    n = B_ * DIN;
    cvt_f32_to_bf16<<<(n / 4 + 255) / 256, 256, 0, stream>>>(x, xb, n);
    n = DHID * DIN;
    cvt_f32_to_bf16<<<(n / 4 + 255) / 256, 256, 0, stream>>>(W1, W1b, n);
    n = NF * DHID;
    cvt_f32_to_bf16<<<(n / 4 + 255) / 256, 256, 0, stream>>>(W2, W2b, n);
  }

  // 2) GEMM1: hid = relu(x @ W1^T + b1) -> bf16  [B, DHID]
  {
    dim3 grid(B_ / 128, DHID / 64);
    gemm_bf16_wmma<1><<<grid, 256, 0, stream>>>(
        xb, W1b, b1, nullptr, nullptr, hidb, nullptr, B_, DHID, DIN);
  }

  // 3) GEMM2: h = (hid @ W2^T + b2)*scale + bias -> fp32  [B, NF]
  {
    dim3 grid(B_ / 128, NF / 64);
    gemm_bf16_wmma<2><<<grid, 256, 0, stream>>>(
        hidb, W2b, b2, scale, bias, nullptr, hbuf, B_, NF, DHID);
  }

  // 4) Wd [DIN, NF] -> WdT [NF, DIN]
  {
    dim3 grid(NF / 32, DIN / 32);
    transpose_kernel<<<grid, 256, 0, stream>>>(Wd, WdT, DIN, NF);
  }

  // 5) top-k + scatter into feats (also compact idx/val)
  {
    const int threads = 512;
    size_t shmem = (size_t)NF * 4 + threads * 8 + TK * 8;
    topk_scatter_kernel<<<B_, threads, shmem, stream>>>(
        hbuf, feats, cidx, cval, NF, TK);
  }

  // 6) sparse decoder -> recon
  sparse_decoder_kernel<<<B_, 256, 0, stream>>>(WdT, bd, cidx, cval, recon, DIN, TK);
}
